// SimpleTokenSelect_12610023981643
// MI455X (gfx1250) — matrix operand order repaired and checked by reading, compiled, not run on hardware
//
#include <hip/hip_runtime.h>
#include <hip/hip_bf16.h>
#include <math.h>

// Problem dims (fixed by the reference)
#define BB 8
#define HH 12
#define LL 1024
#define DD 768

typedef __attribute__((ext_vector_type(16))) _Float16 v16h;
typedef __attribute__((ext_vector_type(8)))  float    v8f;
typedef __attribute__((ext_vector_type(4)))  float    v4f;   // true vector type: OK for nontemporal builtins
typedef __attribute__((ext_vector_type(2)))  float    v2f;

// ---------------------------------------------------------------------------
// Kernel 1: token-select gate via WMMA GEMV.
// One wave (32 lanes) computes logits for 16 tokens using
// v_wmma_f32_16x16x32_f16, K looped over D=768 in steps of 32.
// B-matrix: all 16 columns hold W[c..c+31], so C[:,0] = logits.
// Out-of-range token row (only tile 63 / m=15) is CLAMPED, not guarded:
// loads stay unconditional/coalesced; the bogus result is simply not stored.
// ---------------------------------------------------------------------------
__global__ __launch_bounds__(32)
void token_select_kernel(const float* __restrict__ x,
                         const float* __restrict__ W,
                         const float* __restrict__ bias,
                         float* __restrict__ ts_out)
{
    const int tile = blockIdx.x & 63;          // 64 tiles of 16 tokens
    const int b    = blockIdx.x >> 6;
    const int lane = threadIdx.x;              // 0..31 (wave32)
    const int half_sel = lane >> 4;            // 0: lanes 0-15, 1: lanes 16-31
    const int m    = lane & 15;                // A-matrix row within tile
    const int tok  = 1 + tile * 16 + m;        // tokens 1..1024 (skip CLS)
    const int tokc = (tok < LL) ? tok : (LL - 1);   // clamp -> always in bounds

    const float* __restrict__ xrow = x + ((size_t)b * LL + tokc) * DD;

    v8f acc = {};
    for (int c = 0; c < DD; c += 32) {
        v16h a, bm;
#pragma unroll
        for (int v = 0; v < 8; ++v) {
            // 16-bit A 16x32 layout (ISA 7.12.2):
            // VGPR v<4: lanes0-15 K=2v,2v+1; lanes16-31 K=8+2v,9+2v
            // VGPR v>=4: lanes0-15 K=16+2(v-4)..; lanes16-31 +8 more
            const int kA = ((v < 4) ? (2 * v) : (16 + 2 * (v - 4))) + half_sel * 8;
            const v2f a2 = *reinterpret_cast<const v2f*>(xrow + c + kA);
            a[2 * v]     = (_Float16)a2[0];
            a[2 * v + 1] = (_Float16)a2[1];
            // B 32x16 layout: same K order as A (lanes0-15 K=0..7,16..23; lanes16-31 K=8..15,24..31).
            // All N columns identical = W slice -> every column of C is the logits.
            const int kB = ((v < 4) ? (2 * v) : (16 + 2 * (v - 4))) + half_sel * 8;
            const v2f w2 = *reinterpret_cast<const v2f*>(W + c + kB);
            bm[2 * v]     = (_Float16)w2[0];
            bm[2 * v + 1] = (_Float16)w2[1];
        }
        acc = __builtin_amdgcn_wmma_f32_16x16x32_f16(
                  /*neg_a=*/false, a, /*neg_b=*/false, bm,
                  /*c_mod=*/(short)0, acc, /*reuse_a=*/false, /*reuse_b=*/false);
    }

    // C layout (f32 16x16): VGPR v, lanes 0-15 -> M=v, lanes 16-31 -> M=v+8.
    // Column N=0 lives in lanes 0 and 16.
    if ((lane & 15) == 0) {
        const float bv = bias[0];
#pragma unroll
        for (int v = 0; v < 8; ++v) {
            const int mm = v + half_sel * 8;
            const int t  = 1 + tile * 16 + mm;
            if (t < LL) {
                const float logit  = acc[v] + bv;
                const float y_soft = 1.0f / (1.0f + __expf(-logit));
                const float y_hard = (y_soft > 0.5f) ? 1.0f : 0.0f;
                // straight-through: (y_hard + y_soft) - y_soft, exact fp order
                const float ts = (y_hard + y_soft) - y_soft;
                ts_out[(size_t)b * LL + t] = ts;
            }
        }
        if (tile == 0 && lane == 0) {
            ts_out[(size_t)b * LL] = 1.0f;   // CLS always kept
        }
    }
}

// ---------------------------------------------------------------------------
// Kernel 2: masked streaming softmax (the 806 MB HBM-bound part, ~35us floor
// at 23.3 TB/s). Block = 256 threads = 8 waves; each wave owns one row i of
// length 1024, scores live in 32 VGPRs/lane -> single pass over attn_pre.
// token_select row staged to LDS via CDNA5 async DMA
// (global_load_async_to_lds_b128 + s_wait_asynccnt).
// Main stream uses non-temporal loads/stores: 403MB each way, read/written
// exactly once, > 192MB L2 -> keep it out of the cache.
// ---------------------------------------------------------------------------
__global__ __launch_bounds__(256)
void masked_softmax_kernel(const float* __restrict__ attn_pre,
                           const float* __restrict__ ts,
                           float* __restrict__ out)
{
    const int chunks = LL / 8;                 // 128 row-chunks per (b,h)
    const int rc = blockIdx.x % chunks;
    const int bh = blockIdx.x / chunks;        // 0..B*H-1
    const int b  = bh / HH;

    __shared__ float s_sel[LL];
    const int tid = threadIdx.x;

    // Async DMA: 256 lanes x 16B = 4KB token_select row -> LDS directly.
    {
        // Generic pointers to LDS carry the LDS byte offset in the low 32 bits.
        unsigned lds_off = (unsigned)(size_t)(&s_sel[0]) + (unsigned)tid * 16u;
        unsigned long long gaddr =
            (unsigned long long)(const void*)(ts + (size_t)b * LL) + (size_t)tid * 16u;
        asm volatile("global_load_async_to_lds_b128 %0, %1, off"
                     :: "v"(lds_off), "v"(gaddr) : "memory");
        asm volatile("s_wait_asynccnt 0x0" ::: "memory");
    }
    __syncthreads();

    const int wave = tid >> 5;
    const int lane = tid & 31;
    const int i    = rc * 8 + wave;
    const float s_i = s_sel[i];

    const size_t rowoff = ((size_t)bh * LL + i) * (size_t)LL;
    const v4f* __restrict__ row4  = reinterpret_cast<const v4f*>(attn_pre + rowoff);
    v4f*       __restrict__ orow4 = reinterpret_cast<v4f*>(out + rowoff);
    const v4f* __restrict__ sel4  = reinterpret_cast<const v4f*>(s_sel);

    float sc[32];
    float mx = -INFINITY;
#pragma unroll
    for (int k = 0; k < 8; ++k) {
        const int idx = k * 32 + lane;         // float4 index within the row
        const v4f a4 = __builtin_nontemporal_load(&row4[idx]);
        const v4f s4 = sel4[idx];              // ds_load_b128
        const int j0 = idx * 4;
#pragma unroll
        for (int e = 0; e < 4; ++e) {
            const int j = j0 + e;
            const float p = s_i * s4[e];
            // reference: masked = (1 - policy - eye) > 0 ; scores = a*p + (-inf if masked)
            const float eye = (i == j) ? 1.0f : 0.0f;
            const bool masked = (1.0f - p - eye) > 0.0f;
            const float v = masked ? -INFINITY : a4[e] * p;
            sc[k * 4 + e] = v;
            mx = fmaxf(mx, v);
        }
    }
    // wave32 max reduction
#pragma unroll
    for (int off = 16; off > 0; off >>= 1)
        mx = fmaxf(mx, __shfl_xor(mx, off, 32));

    float sum = 0.0f;
#pragma unroll
    for (int k = 0; k < 32; ++k) {
        const float e = __expf(sc[k] - mx);    // exp(-inf)=0 handles masked lanes
        sc[k] = e;
        sum += e;
    }
#pragma unroll
    for (int off = 16; off > 0; off >>= 1)
        sum += __shfl_xor(sum, off, 32);

    const float inv = 1.0f / sum;
#pragma unroll
    for (int k = 0; k < 8; ++k) {
        v4f o;
        o[0] = sc[k * 4 + 0] * inv;
        o[1] = sc[k * 4 + 1] * inv;
        o[2] = sc[k * 4 + 2] * inv;
        o[3] = sc[k * 4 + 3] * inv;
        __builtin_nontemporal_store(o, &orow4[k * 32 + lane]);
    }
}

// ---------------------------------------------------------------------------
// Launch. d_in order: x, attn (UNUSED by reference), attn_pre_softmax, W, b.
// d_out = attn_out (B*H*L*L floats) ++ token_select (B*L floats).
// ---------------------------------------------------------------------------
extern "C" void kernel_launch(void* const* d_in, const int* in_sizes, int n_in,
                              void* d_out, int out_size, void* d_ws, size_t ws_size,
                              hipStream_t stream)
{
    const float* x        = (const float*)d_in[0];
    // d_in[1] (attn) is never read by the reference
    const float* attn_pre = (const float*)d_in[2];
    const float* W        = (const float*)d_in[3];
    const float* bias     = (const float*)d_in[4];

    float* out = (float*)d_out;
    float* ts  = out + (size_t)BB * HH * LL * LL;   // token_select lives at the tail

    token_select_kernel<<<BB * 64, 32, 0, stream>>>(x, W, bias, ts);
    masked_softmax_kernel<<<BB * HH * (LL / 8), 256, 0, stream>>>(attn_pre, ts, out);
}
